// ImageTransformer2D_3D_Cross_Proj_49435073577559
// MI455X (gfx1250) — compile-verified, matrix operand order checked by reading, not run on hardware
//
#include <hip/hip_runtime.h>
#include <math.h>

// ---------------------------------------------------------------------------
// CDNA5 / gfx1250 implementation of the TransFusion-style image decoder.
// Matrix contractions lower to v_wmma_f32_16x16x32_bf16 (wave32 WMMA).
// Attention V-tiles are staged through LDS with global_load_async_to_lds_b128
// (ASYNCcnt) overlapped with the QK^T WMMA + online softmax.
// ---------------------------------------------------------------------------

typedef __attribute__((ext_vector_type(16))) __bf16 v16bf;
typedef __attribute__((ext_vector_type(8)))  float  v8f;

constexpr int B_ = 2, C_ = 128, NQ = 600, V_ = 6, HW_ = 5600;
constexpr int L_ = 3, F_ = 256, HC_ = 64, H_ = 8, D_ = 16;
constexpr float NEGB = -1e9f;

constexpr size_t S_BNQC = (size_t)B_ * NQ * C_;    // 153600
constexpr size_t S_HWC  = (size_t)HW_ * C_;        // 716800
constexpr size_t S_BHWC = (size_t)B_ * HW_ * C_;   // 1433600

// d_out layout (flat, reference return order)
constexpr size_t OUT_QPOS = (size_t)B_ * C_ * NQ;               // 153600
constexpr size_t OUT_CEN  = OUT_QPOS + (size_t)B_ * NQ * 2;     // 156000
constexpr size_t OUT_OFF  = OUT_CEN + (size_t)L_ * B_ * 2 * NQ; // 163200
constexpr size_t OUT_BOX  = OUT_OFF + (size_t)L_ * B_ * 4 * NQ; // 177600

// 16-bit WMMA fragment K layout (ISA 7.12.2): per lane, element e maps to
// k = ((e&8)<<1) + (half<<3) + (e&7), i.e. two contiguous 8-float runs at
// k0 + half*8 and k0 + half*8 + 16.

__device__ __forceinline__ void pack8(v16bf& d, int base, float4 x, float4 y) {
  d[base + 0] = (__bf16)x.x; d[base + 1] = (__bf16)x.y;
  d[base + 2] = (__bf16)x.z; d[base + 3] = (__bf16)x.w;
  d[base + 4] = (__bf16)y.x; d[base + 5] = (__bf16)y.y;
  d[base + 6] = (__bf16)y.z; d[base + 7] = (__bf16)y.w;
}
__device__ __forceinline__ void pack8s(v16bf& d, int base, float4 x, float4 y,
                                       float s) {
  d[base + 0] = (__bf16)(x.x * s); d[base + 1] = (__bf16)(x.y * s);
  d[base + 2] = (__bf16)(x.z * s); d[base + 3] = (__bf16)(x.w * s);
  d[base + 4] = (__bf16)(y.x * s); d[base + 5] = (__bf16)(y.y * s);
  d[base + 6] = (__bf16)(y.z * s); d[base + 7] = (__bf16)(y.w * s);
}
__device__ __forceinline__ void zero8(v16bf& d, int base) {
#pragma unroll
  for (int e = 0; e < 8; ++e) d[base + e] = (__bf16)0.f;
}

// Async copy of 16 bytes from global to LDS (GV mode, ASYNCcnt-tracked).
__device__ __forceinline__ void async_g2l_b128(unsigned lds_byte_addr,
                                               const float* gaddr) {
  asm volatile("global_load_async_to_lds_b128 %0, %1, off"
               :
               : "v"(lds_byte_addr), "v"(gaddr)
               : "memory");
}
__device__ __forceinline__ void wait_async0() {
  asm volatile("s_wait_asynccnt 0x0" ::: "memory");
}

// ---------------------------------------------------------------------------
// WMMA GEMM: C[m,n] = sum_k A[m,k] * W[n,k] (+bias[n]) (+resid) (relu).
// Requirements: M % 16 == 0, K % 32 == 0, lda/ldw % 4 == 0 (all call sites
// satisfy this). N may be ragged (clamped loads, guarded stores).
// One wave computes a 16(M) x 64(N) tile: A-frag reused across 4 WMMAs.
// ---------------------------------------------------------------------------
__global__ void gemm_bt_wmma(const float* __restrict__ A, int lda,
                             const float* __restrict__ W, int ldw,
                             const float* __restrict__ bias,
                             const float* __restrict__ resid, int ldr,
                             float* __restrict__ Co, int ldc,
                             int M, int N, int K, int relu) {
  const int lane = threadIdx.x & 31;
  const int half = lane >> 4, idx = lane & 15;
  const int tm = blockIdx.y * 16;
  const int tn0 = blockIdx.x * 64;
  const int m = tm + idx;  // always < M
  const float* Arow = A + (size_t)m * lda + half * 8;
  const float* Wrow[4];
#pragma unroll
  for (int t = 0; t < 4; ++t) {
    int n = tn0 + t * 16 + idx;
    int nn = n < N ? n : N - 1;  // clamp: branch-free loads
    Wrow[t] = W + (size_t)nn * ldw + half * 8;
  }
  v8f acc[4] = {{}, {}, {}, {}};
  for (int k0 = 0; k0 < K; k0 += 32) {
    if (k0 + 32 < K) {
      __builtin_prefetch(Arow + k0 + 32, 0, 3);
      __builtin_prefetch(Wrow[0] + k0 + 32, 0, 3);
    }
    v16bf a;
    pack8(a, 0, *(const float4*)(Arow + k0),      *(const float4*)(Arow + k0 + 4));
    pack8(a, 8, *(const float4*)(Arow + k0 + 16), *(const float4*)(Arow + k0 + 20));
#pragma unroll
    for (int t = 0; t < 4; ++t) {
      v16bf b;
      pack8(b, 0, *(const float4*)(Wrow[t] + k0),
                  *(const float4*)(Wrow[t] + k0 + 4));
      pack8(b, 8, *(const float4*)(Wrow[t] + k0 + 16),
                  *(const float4*)(Wrow[t] + k0 + 20));
      acc[t] = __builtin_amdgcn_wmma_f32_16x16x32_bf16(false, a, false, b,
                                                       (short)0, acc[t],
                                                       false, false);
    }
  }
#pragma unroll
  for (int t = 0; t < 4; ++t) {
    int col = tn0 + t * 16 + idx;
    if (col < N) {
      float bi = bias ? bias[col] : 0.f;
#pragma unroll
      for (int r = 0; r < 8; ++r) {
        int row = tm + r + (half << 3);
        float v = acc[t][r] + bi;
        if (resid) v += resid[(size_t)row * ldr + col];
        if (relu)  v = fmaxf(v, 0.f);
        Co[(size_t)row * ldc + col] = v;
      }
    }
  }
}

// ---------------------------------------------------------------------------
// Flash self-attention, mask = (view[q]==view[k] ? 0 : -1e9).
// One wave handles (b, h, 16-query tile); streams keys in chunks of 16.
// QKV: [B*NQ, 3C] with Q|K|V at offsets 0|C|2C, head h at h*16.
// V tile (16 keys x 16 dims) is async-DMAed to LDS, overlapped with QK^T.
// ---------------------------------------------------------------------------
__global__ void attn_self(const float* __restrict__ QKV,
                          const int* __restrict__ view,
                          float* __restrict__ O) {
  __shared__ __align__(16) float pbuf[16][20];
  __shared__ __align__(16) float vtile[256];  // [key][d]
  const int lane = threadIdx.x & 31, half = lane >> 4, idx = lane & 15;
  const int b = blockIdx.z, h = blockIdx.y, tq = blockIdx.x * 16;
  const float scale = 0.25f;  // D^-0.5, D=16
  const unsigned vbase = (unsigned)(uintptr_t)vtile;

  v16bf qa;
  {
    int qi = tq + idx; if (qi >= NQ) qi = NQ - 1;  // clamp; masked later
    const float* Qr = QKV + (size_t)(b * NQ + qi) * (3 * C_) + h * D_ + half * 8;
    pack8s(qa, 0, *(const float4*)(Qr), *(const float4*)(Qr + 4), scale);
    zero8(qa, 8);  // k >= 16 padding
  }
  int qv[8];
#pragma unroll
  for (int r = 0; r < 8; ++r) {
    int q = tq + r + (half << 3);
    qv[r] = (q < NQ) ? view[b * NQ + q] : -2;
  }
  float mrow[8], lrow[8];
  v8f o = {};
#pragma unroll
  for (int r = 0; r < 8; ++r) { mrow[r] = -3.0e38f; lrow[r] = 0.f; }

  for (int kc = 0; kc < NQ; kc += 16) {
    // kick off async V-tile copy to LDS (overlaps with QK^T + softmax below)
#pragma unroll
    for (int j = 0; j < 2; ++j) {
      int lin = lane + j * 32;                 // 64 x 16B chunks -> 1KB tile
      int key = lin >> 2;                      // 0..15
      int d0 = (lin & 3) << 2;                 // 0,4,8,12
      int k2 = kc + key; if (k2 >= NQ) k2 = NQ - 1;  // p==0 for OOB keys
      async_g2l_b128(vbase + (unsigned)(lin * 16),
                     QKV + (size_t)(b * NQ + k2) * (3 * C_) + 2 * C_ + h * D_ + d0);
    }
    int key = kc + idx;
    int ki = key < NQ ? key : NQ - 1;
    int kvw = (key < NQ) ? view[b * NQ + key] : -3;
    v16bf kb;
    {
      const float* Kr =
          QKV + (size_t)(b * NQ + ki) * (3 * C_) + C_ + h * D_ + half * 8;
      pack8(kb, 0, *(const float4*)(Kr), *(const float4*)(Kr + 4));
      zero8(kb, 8);
    }
    v8f s = {};
    s = __builtin_amdgcn_wmma_f32_16x16x32_bf16(false, qa, false, kb,
                                                (short)0, s, false, false);
#pragma unroll
    for (int r = 0; r < 8; ++r) {
      float sv = s[r];
      if (key >= NQ || qv[r] < -1) sv = -3.0e38f;
      else if (kvw != qv[r])       sv += NEGB;
      float cm = sv;
      cm = fmaxf(cm, __shfl_xor(cm, 1));
      cm = fmaxf(cm, __shfl_xor(cm, 2));
      cm = fmaxf(cm, __shfl_xor(cm, 4));
      cm = fmaxf(cm, __shfl_xor(cm, 8));
      float mnew = fmaxf(mrow[r], cm);
      float alpha = __expf(mrow[r] - mnew);
      float p = __expf(sv - mnew);
      float rs = p;
      rs += __shfl_xor(rs, 1); rs += __shfl_xor(rs, 2);
      rs += __shfl_xor(rs, 4); rs += __shfl_xor(rs, 8);
      lrow[r] = lrow[r] * alpha + rs;
      mrow[r] = mnew;
      o[r] *= alpha;
      pbuf[r + (half << 3)][idx] = p;
    }
    __syncthreads();
    v16bf pa, vb;
    pack8(pa, 0, *(const float4*)&pbuf[idx][half * 8],
                 *(const float4*)&pbuf[idx][half * 8 + 4]);
    zero8(pa, 8);
    wait_async0();  // V tile resident in LDS
#pragma unroll
    for (int e = 0; e < 8; ++e)
      vb[e] = (__bf16)vtile[(half * 8 + e) * 16 + idx];
    zero8(vb, 8);
    __syncthreads();
    o = __builtin_amdgcn_wmma_f32_16x16x32_bf16(false, pa, false, vb,
                                                (short)0, o, false, false);
  }
#pragma unroll
  for (int r = 0; r < 8; ++r) {
    int q = tq + r + (half << 3);
    if (q < NQ)
      O[((size_t)(b * NQ + q)) * C_ + h * D_ + idx] = o[r] / lrow[r];
  }
}

// ---------------------------------------------------------------------------
// Flash cross-attention for one view: queries with view==vsel attend over the
// HW tokens of that view. Early-exits tiles with no matching query.
// Qc: [B*NQ, C]; Kc/Vc: [B*HW, C]. HW % 16 == 0.
// ---------------------------------------------------------------------------
__global__ void attn_cross(const float* __restrict__ Qc,
                           const float* __restrict__ Kc,
                           const float* __restrict__ Vc,
                           const int* __restrict__ view, int vsel,
                           float* __restrict__ O) {
  __shared__ __align__(16) float pbuf[16][20];
  __shared__ __align__(16) float vtile[256];  // [key][d]
  const int lane = threadIdx.x & 31, half = lane >> 4, idx = lane & 15;
  const int b = blockIdx.z, h = blockIdx.y, tq = blockIdx.x * 16;

  int q0 = tq + idx;
  int myv = (q0 < NQ) ? view[b * NQ + q0] : -1;
  if (!__any(myv == vsel)) return;  // wave-uniform early exit

  const float scale = 0.25f;
  const unsigned vbase = (unsigned)(uintptr_t)vtile;
  v16bf qa;
  {
    int qi = q0 < NQ ? q0 : NQ - 1;
    const float* Qr = Qc + (size_t)(b * NQ + qi) * C_ + h * D_ + half * 8;
    pack8s(qa, 0, *(const float4*)(Qr), *(const float4*)(Qr + 4), scale);
    zero8(qa, 8);
  }
  float mrow[8], lrow[8];
  v8f o = {};
#pragma unroll
  for (int r = 0; r < 8; ++r) { mrow[r] = -3.0e38f; lrow[r] = 0.f; }

  for (int kc = 0; kc < HW_; kc += 16) {
    // async V-tile DMA to LDS, overlapped with QK^T + softmax
#pragma unroll
    for (int j = 0; j < 2; ++j) {
      int lin = lane + j * 32;
      int key = lin >> 2;
      int d0 = (lin & 3) << 2;
      async_g2l_b128(vbase + (unsigned)(lin * 16),
                     Vc + (size_t)(b * HW_ + kc + key) * C_ + h * D_ + d0);
    }
    v16bf kb;
    {
      const float* Kr = Kc + (size_t)(b * HW_ + kc + idx) * C_ + h * D_ + half * 8;
      pack8(kb, 0, *(const float4*)(Kr), *(const float4*)(Kr + 4));
      zero8(kb, 8);
    }
    v8f s = {};
    s = __builtin_amdgcn_wmma_f32_16x16x32_bf16(false, qa, false, kb,
                                                (short)0, s, false, false);
#pragma unroll
    for (int r = 0; r < 8; ++r) {
      float sv = s[r];
      float cm = sv;
      cm = fmaxf(cm, __shfl_xor(cm, 1));
      cm = fmaxf(cm, __shfl_xor(cm, 2));
      cm = fmaxf(cm, __shfl_xor(cm, 4));
      cm = fmaxf(cm, __shfl_xor(cm, 8));
      float mnew = fmaxf(mrow[r], cm);
      float alpha = __expf(mrow[r] - mnew);
      float p = __expf(sv - mnew);
      float rs = p;
      rs += __shfl_xor(rs, 1); rs += __shfl_xor(rs, 2);
      rs += __shfl_xor(rs, 4); rs += __shfl_xor(rs, 8);
      lrow[r] = lrow[r] * alpha + rs;
      mrow[r] = mnew;
      o[r] *= alpha;
      pbuf[r + (half << 3)][idx] = p;
    }
    __syncthreads();
    v16bf pa, vb;
    pack8(pa, 0, *(const float4*)&pbuf[idx][half * 8],
                 *(const float4*)&pbuf[idx][half * 8 + 4]);
    zero8(pa, 8);
    wait_async0();
#pragma unroll
    for (int e = 0; e < 8; ++e)
      vb[e] = (__bf16)vtile[(half * 8 + e) * 16 + idx];
    zero8(vb, 8);
    __syncthreads();
    o = __builtin_amdgcn_wmma_f32_16x16x32_bf16(false, pa, false, vb,
                                                (short)0, o, false, false);
  }
#pragma unroll
  for (int r = 0; r < 8; ++r) {
    int q = tq + r + (half << 3);
    if (q < NQ && view[b * NQ + q] == vsel)
      O[((size_t)(b * NQ + q)) * C_ + h * D_ + idx] = o[r] / lrow[r];
  }
}

// ---------------------------------------------------------------------------
// Small helper kernels
// ---------------------------------------------------------------------------
__global__ void ln_rows(const float* __restrict__ X, const float* __restrict__ g,
                        const float* __restrict__ bb, float* __restrict__ Y) {
  int row = blockIdx.x;
  int lane = threadIdx.x;
  float4 v = *(const float4*)(X + (size_t)row * C_ + lane * 4);
  float s = v.x + v.y + v.z + v.w;
  float ss = v.x * v.x + v.y * v.y + v.z * v.z + v.w * v.w;
  for (int m = 1; m < 32; m <<= 1) { s += __shfl_xor(s, m); ss += __shfl_xor(ss, m); }
  float mean = s / C_;
  float var = ss / C_ - mean * mean;
  float rstd = rsqrtf(var + 1e-5f);
  float4 gg = *(const float4*)(g + lane * 4);
  float4 bv = *(const float4*)(bb + lane * 4);
  float4 y;
  y.x = (v.x - mean) * rstd * gg.x + bv.x;
  y.y = (v.y - mean) * rstd * gg.y + bv.y;
  y.z = (v.z - mean) * rstd * gg.z + bv.z;
  y.w = (v.w - mean) * rstd * gg.w + bv.w;
  *(float4*)(Y + (size_t)row * C_ + lane * 4) = y;
}

__global__ void posmlp1(const float* __restrict__ pos, const float* __restrict__ w1,
                        const float* __restrict__ b1, float* __restrict__ Hh, int rows) {
  int idx = blockIdx.x * blockDim.x + threadIdx.x;
  if (idx >= rows * C_) return;
  int r = idx / C_, c = idx % C_;
  float v = pos[r * 2 + 0] * w1[c * 2 + 0] + pos[r * 2 + 1] * w1[c * 2 + 1] + b1[c];
  Hh[idx] = fmaxf(v, 0.f);
}

__global__ void addf(const float* __restrict__ a, const float* __restrict__ b,
                     float* __restrict__ c, size_t n) {
  size_t i = (size_t)blockIdx.x * blockDim.x + threadIdx.x;
  if (i < n) c[i] = a[i] + b[i];
}

__global__ void copyf(const float* __restrict__ a, float* __restrict__ b, size_t n) {
  size_t i = (size_t)blockIdx.x * blockDim.x + threadIdx.x;
  if (i < n) b[i] = a[i];
}

__global__ void make_kk(const float* __restrict__ feat, const float* __restrict__ kpe,
                        float* __restrict__ kk, int v) {
  size_t idx = (size_t)blockIdx.x * blockDim.x + threadIdx.x;
  if (idx >= S_BHWC) return;
  int b = (int)(idx / ((size_t)HW_ * C_));
  int r = (int)(idx % ((size_t)HW_ * C_));
  int hw = r / C_, c = r % C_;
  kk[idx] = feat[(((size_t)b * V_ + v) * C_ + c) * HW_ + hw] + kpe[(size_t)hw * C_ + c];
}

__global__ void feat_in(const float* __restrict__ f, float* __restrict__ x) {
  size_t idx = (size_t)blockIdx.x * blockDim.x + threadIdx.x;
  if (idx >= S_BNQC) return;
  int b = (int)(idx / ((size_t)NQ * C_));
  int r = (int)(idx % ((size_t)NQ * C_));
  int q = r / C_, c = r % C_;
  x[idx] = f[((size_t)b * C_ + c) * NQ + q];
}

__global__ void feat_out(const float* __restrict__ x, float* __restrict__ out) {
  size_t idx = (size_t)blockIdx.x * blockDim.x + threadIdx.x;
  if (idx >= (size_t)B_ * C_ * NQ) return;
  int b = (int)(idx / ((size_t)C_ * NQ));
  int r = (int)(idx % ((size_t)C_ * NQ));
  int c = r / NQ, q = r % NQ;
  out[idx] = x[((size_t)b * NQ + q) * C_ + c];
}

__global__ void count_views(const int* __restrict__ view, int* __restrict__ cnt) {
  __shared__ int sc[B_ * V_];
  if (threadIdx.x < B_ * V_) sc[threadIdx.x] = 0;
  __syncthreads();
  for (int idx = threadIdx.x; idx < B_ * NQ; idx += blockDim.x) {
    int b = idx / NQ;
    int v = view[idx];
    if (v >= 0 && v < V_) atomicAdd(&sc[b * V_ + v], 1);
  }
  __syncthreads();
  if (threadIdx.x < B_ * V_) cnt[threadIdx.x] = sc[threadIdx.x];
}

__global__ void apply_keep(const float* __restrict__ t, const int* __restrict__ view,
                           const int* __restrict__ cnt, float* __restrict__ x) {
  size_t idx = (size_t)blockIdx.x * blockDim.x + threadIdx.x;
  if (idx >= S_BNQC) return;
  int row = (int)(idx / C_);
  int b = row / NQ;
  int vv = view[row];
  x[idx] = (cnt[b * V_ + vv] > 1) ? t[idx] : 0.f;
}

__global__ void head_final(const float* __restrict__ ct, const float* __restrict__ ot,
                           float* __restrict__ qpos, float* __restrict__ out, int layer) {
  int idx = blockIdx.x * blockDim.x + threadIdx.x;
  if (idx >= B_ * NQ) return;
  int b = idx / NQ, q = idx % NQ;
  float c0 = ct[idx * 2 + 0] + qpos[idx * 2 + 0];
  float c1 = ct[idx * 2 + 1] + qpos[idx * 2 + 1];
  qpos[idx * 2 + 0] = c0;
  qpos[idx * 2 + 1] = c1;
  float cs0 = 1.f / (1.f + __expf(-c0));
  float cs1 = 1.f / (1.f + __expf(-c1));
  float os[4];
#pragma unroll
  for (int k = 0; k < 4; ++k) os[k] = 1.f / (1.f + __expf(-ot[idx * 4 + k]));
  float* cen = out + OUT_CEN + (size_t)layer * B_ * 2 * NQ;
  cen[(b * 2 + 0) * NQ + q] = cs0;
  cen[(b * 2 + 1) * NQ + q] = cs1;
  float* off = out + OUT_OFF + (size_t)layer * B_ * 4 * NQ;
#pragma unroll
  for (int k = 0; k < 4; ++k) off[(b * 4 + k) * NQ + q] = os[k];
  float bw = os[0] + os[2], bh = os[1] + os[3];
  float cx = cs0 + 0.5f * (os[2] - os[0]);
  float cy = cs1 + 0.5f * (os[3] - os[1]);
  float* bx = out + OUT_BOX + (size_t)layer * B_ * 4 * NQ;
  bx[(b * 4 + 0) * NQ + q] = cx;
  bx[(b * 4 + 1) * NQ + q] = cy;
  bx[(b * 4 + 2) * NQ + q] = bw;
  bx[(b * 4 + 3) * NQ + q] = bh;
}

// ---------------------------------------------------------------------------
// Host orchestration
// ---------------------------------------------------------------------------
static inline void gemm(hipStream_t s, const float* A, int lda, const float* W,
                        int ldw, const float* bias, const float* resid, int ldr,
                        float* Cc, int ldc, int M, int N, int K, int relu) {
  dim3 g((N + 63) / 64, M / 16);  // all call sites have M % 16 == 0
  gemm_bt_wmma<<<g, dim3(32), 0, s>>>(A, lda, W, ldw, bias, resid, ldr, Cc, ldc,
                                      M, N, K, relu);
}

static inline dim3 eg(size_t n) { return dim3((unsigned)((n + 255) / 256)); }

extern "C" void kernel_launch(void* const* d_in, const int* in_sizes, int n_in,
                              void* d_out, int out_size, void* d_ws, size_t ws_size,
                              hipStream_t stream) {
  (void)in_sizes; (void)n_in; (void)out_size; (void)ws_size;
  const float* img_query_feat = (const float*)d_in[0];   // [B,C,NQ]
  const float* query_pos0     = (const float*)d_in[1];   // [B,NQ,2]
  const int*   view           = (const int*)d_in[2];     // [B,NQ]
  const float* img_feat       = (const float*)d_in[3];   // [B,V,C,HW]
  const float* feat_pos       = (const float*)d_in[4];   // [1,HW,2]
  auto P = [&](int i) { return (const float*)d_in[i]; };

  float* w = (float*)d_ws;
  size_t off = 0;
  auto alloc = [&](size_t n) { size_t r = off; off += n; return r; };
  float* qpos  = w + alloc((size_t)B_ * NQ * 2);
  int*   cnt   = (int*)(w + alloc(16));
  float* qpe   = w + alloc(S_BNQC);
  float* kpe   = w + alloc(S_HWC);
  float* phq   = w + alloc(S_BNQC);   // pos-MLP hidden (queries)
  float* phk   = w + alloc(S_HWC);    // pos-MLP hidden (tokens)
  float* x     = w + alloc(S_BNQC);   // current residual stream [B,NQ,C]
  float* t0    = w + alloc(S_BNQC);
  float* t1    = w + alloc(S_BNQC);   // attention output
  float* t2    = w + alloc(S_BNQC);   // pre-LN buffer
  float* qkv   = w + alloc((size_t)B_ * NQ * 3 * C_);
  float* qc    = w + alloc(S_BNQC);   // cross-attn Q projection
  float* hffn  = w + alloc((size_t)B_ * NQ * F_);
  float* kk    = w + alloc(S_BHWC);   // per-view key input (key + kpe)
  float* Kc    = w + alloc(S_BHWC);
  float* Vc    = w + alloc(S_BHWC);
  float* hh    = w + alloc((size_t)B_ * NQ * HC_);
  float* ct    = w + alloc((size_t)B_ * NQ * 2);
  float* ot    = w + alloc((size_t)B_ * NQ * 4);

  const int MQ = B_ * NQ;           // 1200
  const int MKV = B_ * HW_;         // 11200
  float* out = (float*)d_out;

  // static setup
  copyf<<<eg(MQ * 2), 256, 0, stream>>>(query_pos0, qpos, (size_t)MQ * 2);
  count_views<<<1, 256, 0, stream>>>(view, cnt);
  feat_in<<<eg(S_BNQC), 256, 0, stream>>>(img_query_feat, x);

  for (int i = 0; i < L_; ++i) {
    const float* sa_in_w  = P(5)  + (size_t)i * 3 * C_ * C_;
    const float* sa_in_b  = P(6)  + (size_t)i * 3 * C_;
    const float* sa_out_w = P(7)  + (size_t)i * C_ * C_;
    const float* sa_out_b = P(8)  + (size_t)i * C_;
    const float* ca_in_w  = P(9)  + (size_t)i * 3 * C_ * C_;
    const float* ca_in_b  = P(10) + (size_t)i * 3 * C_;
    const float* ca_out_w = P(11) + (size_t)i * C_ * C_;
    const float* ca_out_b = P(12) + (size_t)i * C_;
    const float* ffn_w1   = P(13) + (size_t)i * F_ * C_;
    const float* ffn_b1   = P(14) + (size_t)i * F_;
    const float* ffn_w2   = P(15) + (size_t)i * C_ * F_;
    const float* ffn_b2   = P(16) + (size_t)i * C_;
    const float* ln_g     = P(17) + (size_t)i * 3 * C_;
    const float* ln_b     = P(18) + (size_t)i * 3 * C_;
    const float* qpos_w1  = P(19) + (size_t)i * C_ * 2;
    const float* qpos_b1  = P(20) + (size_t)i * C_;
    const float* qpos_w2  = P(21) + (size_t)i * C_ * C_;
    const float* qpos_b2  = P(22) + (size_t)i * C_;
    const float* kpos_w1  = P(23) + (size_t)i * C_ * 2;
    const float* kpos_b1  = P(24) + (size_t)i * C_;
    const float* kpos_w2  = P(25) + (size_t)i * C_ * C_;
    const float* kpos_b2  = P(26) + (size_t)i * C_;

    // position embeddings (stage 1: K=2 VALU; stage 2: 128x128 WMMA GEMM)
    posmlp1<<<eg((size_t)MQ * C_), 256, 0, stream>>>(qpos, qpos_w1, qpos_b1, phq, MQ);
    gemm(stream, phq, C_, qpos_w2, C_, qpos_b2, nullptr, 0, qpe, C_, MQ, C_, C_, 0);
    posmlp1<<<eg(S_HWC), 256, 0, stream>>>(feat_pos, kpos_w1, kpos_b1, phk, HW_);
    gemm(stream, phk, C_, kpos_w2, C_, kpos_b2, nullptr, 0, kpe, C_, HW_, C_, C_, 0);

    // ---- self attention (view-masked) ----
    addf<<<eg(S_BNQC), 256, 0, stream>>>(x, qpe, t0, S_BNQC);
    gemm(stream, t0, C_, sa_in_w, C_, sa_in_b, nullptr, 0, qkv, 3 * C_, MQ, 3 * C_, C_, 0);
    attn_self<<<dim3((NQ + 15) / 16, H_, B_), 32, 0, stream>>>(qkv, view, t1);
    gemm(stream, t1, C_, sa_out_w, C_, sa_out_b, x, C_, t2, C_, MQ, C_, C_, 0);
    ln_rows<<<MQ, 32, 0, stream>>>(t2, ln_g + 0 * C_, ln_b + 0 * C_, x);  // x = x1

    // ---- cross attention (per-view K/V, flash) ----
    addf<<<eg(S_BNQC), 256, 0, stream>>>(x, qpe, t0, S_BNQC);
    gemm(stream, t0, C_, ca_in_w, C_, ca_in_b, nullptr, 0, qc, C_, MQ, C_, C_, 0);
    for (int v = 0; v < V_; ++v) {
      make_kk<<<eg(S_BHWC), 256, 0, stream>>>(img_feat, kpe, kk, v);
      gemm(stream, kk, C_, ca_in_w + (size_t)C_ * C_,     C_, ca_in_b + C_,
           nullptr, 0, Kc, C_, MKV, C_, C_, 0);
      gemm(stream, kk, C_, ca_in_w + (size_t)2 * C_ * C_, C_, ca_in_b + 2 * C_,
           nullptr, 0, Vc, C_, MKV, C_, C_, 0);
      attn_cross<<<dim3((NQ + 15) / 16, H_, B_), 32, 0, stream>>>(qc, Kc, Vc, view, v, t1);
    }
    gemm(stream, t1, C_, ca_out_w, C_, ca_out_b, x, C_, t2, C_, MQ, C_, C_, 0);
    ln_rows<<<MQ, 32, 0, stream>>>(t2, ln_g + 1 * C_, ln_b + 1 * C_, x);  // x = x2

    // ---- FFN ----
    gemm(stream, x, C_, ffn_w1, C_, ffn_b1, nullptr, 0, hffn, F_, MQ, F_, C_, 1);
    gemm(stream, hffn, F_, ffn_w2, F_, ffn_b2, x, C_, t2, C_, MQ, C_, F_, 0);
    ln_rows<<<MQ, 32, 0, stream>>>(t2, ln_g + 2 * C_, ln_b + 2 * C_, t0);

    // keep only queries whose view group has >1 member
    apply_keep<<<eg(S_BNQC), 256, 0, stream>>>(t0, view, cnt, x);

    // ---- heads ----
    gemm(stream, x, C_, P(27), C_, P(28), nullptr, 0, hh, HC_, MQ, HC_, C_, 1);
    gemm(stream, hh, HC_, P(29), HC_, P(30), nullptr, 0, ct, 2, MQ, 2, HC_, 0);
    gemm(stream, x, C_, P(31), C_, P(32), nullptr, 0, hh, HC_, MQ, HC_, C_, 1);
    gemm(stream, hh, HC_, P(33), HC_, P(34), nullptr, 0, ot, 4, MQ, 4, HC_, 0);
    head_final<<<eg(MQ), 256, 0, stream>>>(ct, ot, qpos, out, i);
  }

  feat_out<<<eg((size_t)B_ * C_ * NQ), 256, 0, stream>>>(x, out);
  copyf<<<eg((size_t)MQ * 2), 256, 0, stream>>>(qpos, out + OUT_QPOS, (size_t)MQ * 2);
}
